// ImputationModule_59708635349351
// MI455X (gfx1250) — matrix operand order repaired and checked by reading, not compiled, not run on hardware
//
#include <hip/hip_runtime.h>
#include <hip/hip_bf16.h>

// ---------------------------------------------------------------------------
// Problem constants (from reference)
// ---------------------------------------------------------------------------
#define D_M   1024
#define L_OBS 2048
#define ALPHA 8192
#define D_H   2048

typedef __attribute__((ext_vector_type(16))) __bf16 v16bf;
typedef __attribute__((ext_vector_type(8)))  __bf16 bf16x8;
typedef __attribute__((ext_vector_type(8)))  float  v8f;

// ---------------------------------------------------------------------------
// Kernel 1: scatter + forward-fill one m-row per workgroup.
// LDS grid of u64 keys: ((j+1)<<32) | bits(x). atomicMax => latest valid
// observation index wins (matches "later writes win" loop semantics).
// ---------------------------------------------------------------------------
__global__ __launch_bounds__(256)
void fill_rows(const float* __restrict__ x_ts,   // [D_M][L_OBS]
               const int*   __restrict__ t_ts,   // [D_M][L_OBS]
               const float* __restrict__ gmean,  // [D_M]
               __bf16*      __restrict__ regular) // [D_M][ALPHA] bf16 out
{
    __shared__ unsigned long long grid[ALPHA];   // 64 KB
    __shared__ float lastVal[256];
    __shared__ int   lastOk[256];
    __shared__ float carry[256];

    const int m   = blockIdx.x;
    const int tid = threadIdx.x;

    for (int i = tid; i < ALPHA; i += 256) grid[i] = 0ull;
    __syncthreads();

    const float* xr = x_ts + (size_t)m * L_OBS;
    const int*   tr = t_ts + (size_t)m * L_OBS;
    for (int j = tid; j < L_OBS; j += 256) {
        float x = xr[j];
        int   t = tr[j];
        bool valid = (x == x) && (t >= 0) && (t < ALPHA);
        if (valid) {
            unsigned long long key =
                (((unsigned long long)(unsigned)(j + 1)) << 32) |
                (unsigned long long)__float_as_uint(x);
            atomicMax(&grid[t], key);   // -> ds_max_u64 on LDS
        }
    }
    __syncthreads();

    // Phase A: per-thread chunk of 32 grid entries -> last valid value
    const int base = tid * 32;
    float lv = 0.0f; int ok = 0;
    #pragma unroll 4
    for (int i = 0; i < 32; ++i) {
        unsigned long long g = grid[base + i];
        if ((unsigned)(g >> 32) != 0u) { lv = __uint_as_float((unsigned)g); ok = 1; }
    }
    lastVal[tid] = lv;
    lastOk[tid]  = ok;
    __syncthreads();

    // Phase B: serial scan over 256 chunk summaries (cheap)
    if (tid == 0) {
        float cur = gmean[m];
        for (int t = 0; t < 256; ++t) {
            carry[t] = cur;
            if (lastOk[t]) cur = lastVal[t];
        }
    }
    __syncthreads();

    // Phase C: forward-fill chunk, convert to bf16, vector stores
    float cur = carry[tid];
    __align__(16) __bf16 tmp[32];
    #pragma unroll
    for (int i = 0; i < 32; ++i) {
        unsigned long long g = grid[base + i];
        if ((unsigned)(g >> 32) != 0u) cur = __uint_as_float((unsigned)g);
        tmp[i] = (__bf16)cur;
    }
    uint4* dst = (uint4*)(regular + (size_t)m * ALPHA + base);
    const uint4* src = (const uint4*)tmp;
    #pragma unroll
    for (int v = 0; v < 4; ++v) dst[v] = src[v];
}

// ---------------------------------------------------------------------------
// Kernel 2: transpose regular [D_M][ALPHA] -> regT [ALPHA][D_M] (bf16)
// so GEMM A-tiles are row-major along K (= m).
// ---------------------------------------------------------------------------
__global__ __launch_bounds__(256)
void transpose_bf(const __bf16* __restrict__ in,   // [D_M][ALPHA]
                  __bf16*       __restrict__ out)  // [ALPHA][D_M]
{
    __shared__ __bf16 tile[64][64 + 4];
    const int a0  = blockIdx.x * 64;
    const int m0  = blockIdx.y * 64;
    const int tid = threadIdx.x;

    #pragma unroll
    for (int i = 0; i < 16; ++i) {
        int e = tid + i * 256;
        int r = e >> 6, c = e & 63;
        tile[r][c] = in[(size_t)(m0 + r) * ALPHA + a0 + c];
    }
    __syncthreads();
    #pragma unroll
    for (int i = 0; i < 16; ++i) {
        int e = tid + i * 256;
        int r = e >> 6, c = e & 63;
        out[(size_t)(a0 + r) * D_M + m0 + c] = tile[c][r];
    }
}

// ---------------------------------------------------------------------------
// Kernel 3: W f32 [D_H][D_M] -> bf16, same layout ([n][k] == B-frag friendly)
// ---------------------------------------------------------------------------
__global__ __launch_bounds__(256)
void convert_w(const float* __restrict__ W, __bf16* __restrict__ Wb, int n)
{
    int i = blockIdx.x * 256 + threadIdx.x;
    if (i < n) Wb[i] = (__bf16)W[i];
}

// ---------------------------------------------------------------------------
// Kernel 4: GEMM  out[a][h] = sum_k A[a][k]*B[h][k] + bias[h]
// A = regT [ALPHA][D_M] bf16, B = Wb [D_H][D_M] bf16, f32 accumulate in WMMA.
// Block: 256 threads (8 waves), 128x128 output tile, wave grid 2(a) x 4(h),
// each wave: 64x32 = 4x2 v_wmma_f32_16x16x32_bf16 accumulators.
// Staging: GLOBAL_LOAD_ASYNC_TO_LDS_B128 (ASYNCcnt) + double-buffered LDS:
// DMA of tile t+1 overlaps the 16 WMMAs of tile t; one barrier per tile.
// ---------------------------------------------------------------------------
#define KT     64
#define PITCH  72   // 64 + 8 halves padding; rows stay 16B aligned
#define NKT    (D_M / KT)

__device__ __forceinline__ unsigned lds_off_of(const void* p)
{
    // generic LDS pointer = {shared aperture, 32-bit LDS offset}
    return (unsigned)(unsigned long long)p;
}

__device__ __forceinline__ void async_b128_to_lds(unsigned lds_byte, const void* gaddr)
{
    asm volatile("global_load_async_to_lds_b128 %0, %1, off"
                 :: "v"(lds_byte), "v"(gaddr)
                 : "memory");
}

__global__ __launch_bounds__(256)
void gemm_kernel(const __bf16* __restrict__ A,    // [ALPHA][D_M]
                 const __bf16* __restrict__ B,    // [D_H][D_M]
                 const float*  __restrict__ bias, // [D_H]
                 float*        __restrict__ out)  // [ALPHA][D_H]
{
    __shared__ __align__(16) __bf16 As[2][128 * PITCH];
    __shared__ __align__(16) __bf16 Bs[2][128 * PITCH];

    const int a0   = blockIdx.x * 128;
    const int h0   = blockIdx.y * 128;
    const int tid  = threadIdx.x;
    const int lane = tid & 31;
    const int w    = tid >> 5;
    const int wa   = w >> 2;    // 0..1 : a sub-block (64 rows)
    const int wh   = w & 3;     // 0..3 : h sub-block (32 cols)
    const int hi   = lane >> 4; // half-wave select
    const int l16  = lane & 15;

    // Accumulators initialized with bias[h] (C/D layout: N = lane%16)
    v8f acc[4][2];
    #pragma unroll
    for (int tb = 0; tb < 2; ++tb) {
        float bv = bias[h0 + wh * 32 + tb * 16 + l16];
        #pragma unroll
        for (int ta = 0; ta < 4; ++ta)
            #pragma unroll
            for (int r = 0; r < 8; ++r) acc[ta][tb][r] = bv;
    }

    // Staging map: each thread DMAs 16B per pass, 4 passes -> 128x64 halves
    const int srow = tid >> 3;        // 0..31
    const int scol = (tid & 7) * 8;   // halves, 16B granules

    const unsigned ldsA[2] = { lds_off_of(&As[0][0]), lds_off_of(&As[1][0]) };
    const unsigned ldsB[2] = { lds_off_of(&Bs[0][0]), lds_off_of(&Bs[1][0]) };
    const unsigned stByte  = (unsigned)(srow * PITCH + scol) * 2u;

    // issue staging for one K-tile into buffer `buf`
    auto stage = [&](int k0, int buf) {
        #pragma unroll
        for (int p = 0; p < 4; ++p) {
            int r = p * 32;
            async_b128_to_lds(ldsA[buf] + stByte + (unsigned)(r * PITCH) * 2u,
                              A + (size_t)(a0 + srow + r) * D_M + k0 + scol);
            async_b128_to_lds(ldsB[buf] + stByte + (unsigned)(r * PITCH) * 2u,
                              B + (size_t)(h0 + srow + r) * D_M + k0 + scol);
        }
    };

    stage(0, 0);   // prologue

    for (int t = 0; t < NKT; ++t) {
        const int cur = t & 1;
        // my asyncs for buf[cur] (the only outstanding ones) are done:
        asm volatile("s_wait_asynccnt 0" ::: "memory");
        // everyone's buf[cur] staged AND everyone finished compute(t-1),
        // which read buf[cur^1] -> safe to restage it now:
        __syncthreads();
        if (t + 1 < NKT) stage((t + 1) * KT, cur ^ 1);

        const __bf16* AsP = As[cur];
        const __bf16* BsP = Bs[cur];
        #pragma unroll
        for (int kk = 0; kk < KT; kk += 32) {
            union { v16bf v; bf16x8 h[2]; } af[4], bg[2];
            // A frag (ISA 16-bit A 16x32): lane<16 -> K {0..7, 16..23},
            // lane>=16 -> K {8..15, 24..31}; chunks at base, base+16
            const int aK = kk + hi * 8;
            #pragma unroll
            for (int ta = 0; ta < 4; ++ta) {
                const __bf16* p = AsP + (wa * 64 + ta * 16 + l16) * PITCH + aK;
                af[ta].h[0] = *(const bf16x8*)(p);
                af[ta].h[1] = *(const bf16x8*)(p + 16);
            }
            // B frag (32x16): lane<16 -> K {0..7, 16..23}, lane>=16 -> K {8..15, 24..31},
            // N = lane%16; chunks at base, base+16
            const int bK = kk + hi * 8;
            #pragma unroll
            for (int tb = 0; tb < 2; ++tb) {
                const __bf16* p = BsP + (wh * 32 + tb * 16 + l16) * PITCH + bK;
                bg[tb].h[0] = *(const bf16x8*)(p);
                bg[tb].h[1] = *(const bf16x8*)(p + 16);
            }
            #pragma unroll
            for (int ta = 0; ta < 4; ++ta)
                #pragma unroll
                for (int tb = 0; tb < 2; ++tb)
                    acc[ta][tb] = __builtin_amdgcn_wmma_f32_16x16x32_bf16(
                        false, af[ta].v, false, bg[tb].v,
                        (short)0, acc[ta][tb], false, false);
        }
    }

    // Store: VGPR r, lane<16 -> M=r, lane>=16 -> M=r+8; N = lane%16
    #pragma unroll
    for (int ta = 0; ta < 4; ++ta) {
        const int aBase = a0 + wa * 64 + ta * 16 + hi * 8;
        #pragma unroll
        for (int tb = 0; tb < 2; ++tb) {
            const int h = h0 + wh * 32 + tb * 16 + l16;
            #pragma unroll
            for (int r = 0; r < 8; ++r)
                out[(size_t)(aBase + r) * D_H + h] = acc[ta][tb][r];
        }
    }
}

// ---------------------------------------------------------------------------
// Launcher
// ---------------------------------------------------------------------------
extern "C" void kernel_launch(void* const* d_in, const int* in_sizes, int n_in,
                              void* d_out, int out_size, void* d_ws, size_t ws_size,
                              hipStream_t stream)
{
    const float* x_ts  = (const float*)d_in[0];
    const int*   t_ts  = (const int*)  d_in[1];
    const float* gmean = (const float*)d_in[2];
    const float* W     = (const float*)d_in[3];
    const float* b     = (const float*)d_in[4];
    float*       out   = (float*)d_out;

    char* ws = (char*)d_ws;
    __bf16* regular = (__bf16*)(ws);                          // 16 MB [D_M][ALPHA]
    __bf16* regT    = (__bf16*)(ws + ((size_t)16 << 20));     // 16 MB [ALPHA][D_M]
    __bf16* Wb      = (__bf16*)(ws + ((size_t)32 << 20));     //  4 MB [D_H][D_M]

    fill_rows<<<D_M, 256, 0, stream>>>(x_ts, t_ts, gmean, regular);

    dim3 tg(ALPHA / 64, D_M / 64);
    transpose_bf<<<tg, 256, 0, stream>>>(regular, regT);

    convert_w<<<(D_H * D_M) / 256, 256, 0, stream>>>(W, Wb, D_H * D_M);

    dim3 gg(ALPHA / 128, D_H / 128);
    gemm_kernel<<<gg, 256, 0, stream>>>(regT, Wb, b, out);
}
